// deformableResidualblock_49314814493262
// MI455X (gfx1250) — matrix operand order repaired and checked by reading, not compiled, not run on hardware
//
#include <hip/hip_runtime.h>

#ifndef __has_builtin
#define __has_builtin(x) 0
#endif

#define C_CH 96
#define KKONST 9
#define HH 128
#define WW2 128
#define HW 16384
#define BB 4
#define KTOT (C_CH * KKONST)   // 864
#define NEG_SLOPE 0.1f
#define EPS_BN 1e-5f

#define TILE_N 64
#define KSTEP 32
#define WAVES 6
#define TPB (WAVES * 32)
#define NELEM 11               // max B-tile elements per thread (deform kernel)

// Tensor Data Mover path (guarded: arity differs between toolchains)
#if defined(__HIP_DEVICE_COMPILE__) && \
    __has_builtin(__builtin_amdgcn_tensor_load_to_lds) && \
    __has_builtin(__builtin_amdgcn_s_wait_tensorcnt)
#define USE_TDM 1
#if __has_include(<hip/amd_detail/amd_gfx1250_TDM.h>)
#define TDM_6ARG 1
#else
#define TDM_6ARG 0
#endif
#else
#define USE_TDM 0
#endif

typedef __attribute__((ext_vector_type(16))) __bf16          v16bf;
typedef __attribute__((ext_vector_type(16))) unsigned short  v16us;
typedef __attribute__((ext_vector_type(8)))  float           v8f;
typedef __attribute__((ext_vector_type(4)))  unsigned int    v4u;
typedef __attribute__((ext_vector_type(8)))  int             v8i;
typedef __attribute__((ext_vector_type(4)))  int             v4i;

__device__ __forceinline__ unsigned short f32_to_bf16(float f) {
  unsigned int u = __float_as_uint(f);
  u += 0x7FFFu + ((u >> 16) & 1u);   // round-to-nearest-even
  return (unsigned short)(u >> 16);
}

// ---------------------------------------------------------------- weights -> bf16
__global__ void pack_bf16_kernel(const float* __restrict__ in,
                                 unsigned short* __restrict__ out, int n) {
  int i = blockIdx.x * blockDim.x + threadIdx.x;
  if (i < n) out[i] = f32_to_bf16(in[i]);
}

// Concatenate offset(18) + mask(9) conv weights into one 32x864 bf16 A matrix
// (rows 27..31 zero).  Flat OIHW of (O,96,3,3) is already [o][k=c*9+kk].
__global__ void pack_offmask_w_kernel(const float* __restrict__ offw,
                                      const float* __restrict__ modw,
                                      unsigned short* __restrict__ out) {
  int i = blockIdx.x * blockDim.x + threadIdx.x;   // 32*864
  if (i >= 32 * KTOT) return;
  int o = i / KTOT;
  int k = i - o * KTOT;
  float v = 0.f;
  if (o < 18)       v = offw[(size_t)o * KTOT + k];
  else if (o < 27)  v = modw[(size_t)(o - 18) * KTOT + k];
  out[i] = f32_to_bf16(v);
}

// -------- offset+mask conv as WMMA GEMM: M=32(27), K=864, N=64 px / block -------
// 4 waves; wave w owns N-block w (16 px) and both 16-row M-blocks.
// Register-pipelined, double-buffered LDS tiles, one barrier per K-step.
__global__ __launch_bounds__(128) void offmask_gemm_wmma_kernel(
    const float* __restrict__ src, const unsigned short* __restrict__ Aom,
    const float* __restrict__ off_b, const float* __restrict__ mod_b,
    float* __restrict__ off_out, float* __restrict__ mask_out) {
  __shared__ unsigned short As[2][32][40];
  __shared__ unsigned short Bs[2][TILE_N][48];

  const int tid   = threadIdx.x;
  const int lane  = tid & 31;
  const int wave  = tid >> 5;         // 0..3 = N-block
  const int half  = lane >> 4;
  const int lan16 = lane & 15;
  const int j     = tid & 63;         // fixed pixel per thread
  const int klb   = tid >> 6;         // fixed kl residue (0/1), stride 2

  const int b    = blockIdx.x >> 8;
  const int tile = blockIdx.x & 255;
  const int hw0  = tile * TILE_N;
  const int h    = hw0 >> 7;
  const int w0   = hw0 & (WW2 - 1);
  const size_t xbatch = ((size_t)b * C_CH) << 14;

  auto stageA = [&](int bufi, int kb) {
    // 32x32 bf16 = 256 8-byte chunks, 2 per thread
    for (int i = tid; i < 256; i += 128) {
      int m  = i >> 3;
      int kq = (i & 7) * 4;
      *(unsigned long long*)&As[bufi][m][kq] =
          *(const unsigned long long*)&Aom[(size_t)m * KTOT + kb + kq];
    }
  };
  auto gatherB = [&](int kb, float* vbuf) {
    int k  = kb + klb;
    int c  = k / 9;                    // one division per tile
    int kk = k - 9 * c;
    const float* base = src + xbatch;
#pragma unroll
    for (int i = 0; i < 16; ++i) {     // kl = 2*i + klb
      int ky = (kk >= 6) ? 2 : ((kk >= 3) ? 1 : 0);
      int kx = kk - 3 * ky;
      int yy = h - 1 + ky;
      int xx = w0 + j - 1 + kx;
      bool ok = (yy >= 0) && (yy < HH) && (xx >= 0) && (xx < WW2);
      vbuf[i] = ok ? base[((size_t)c << 14) + yy * WW2 + xx] : 0.f;
      kk += 2; if (kk >= 9) { kk -= 9; ++c; }
    }
  };
  auto storeB = [&](int bufi, const float* vbuf) {
#pragma unroll
    for (int i = 0; i < 16; ++i)
      Bs[bufi][j][2 * i + klb] = f32_to_bf16(vbuf[i]);
  };

  // prologue
  float vbuf[16];
  stageA(0, 0);
  gatherB(0, vbuf);
  storeB(0, vbuf);
  __syncthreads();

  v8f acc[2] = {};
  int cur = 0;
  for (int k0 = 0; k0 < KTOT; k0 += KSTEP) {
    const int nxt = cur ^ 1;
    const int knext = k0 + KSTEP;
    const bool hn = knext < KTOT;

    if (hn) gatherB(knext, vbuf);      // vmem overlaps WMMA below

    v16us braw;
    int n = wave * 16 + lan16;
#pragma unroll
    for (int i = 0; i < 16; ++i) braw[i] = Bs[cur][n][((i >> 3) << 4) + half * 8 + (i & 7)];
    v16bf bfrag = __builtin_bit_cast(v16bf, braw);
#pragma unroll
    for (int mb = 0; mb < 2; ++mb) {
      v16us araw;
      int m = mb * 16 + lan16;
#pragma unroll
      for (int i = 0; i < 8; ++i) {
        araw[i]     = As[cur][m][half * 8 + i];
        araw[8 + i] = As[cur][m][16 + half * 8 + i];
      }
      v16bf afrag = __builtin_bit_cast(v16bf, araw);
      acc[mb] = __builtin_amdgcn_wmma_f32_16x16x32_bf16(
          false, afrag, false, bfrag, (short)0, acc[mb], false, false);
    }

    if (hn) { stageA(nxt, knext); storeB(nxt, vbuf); }
    __syncthreads();
    cur = nxt;
  }

  int hw = hw0 + wave * 16 + lan16;
#pragma unroll
  for (int mb = 0; mb < 2; ++mb)
#pragma unroll
    for (int r = 0; r < 8; ++r) {
      int ch = mb * 16 + r + (half << 3);
      float v = acc[mb][r];
      if (ch < 18) {
        off_out[(((size_t)(b * 18 + ch)) << 14) + hw] = v + off_b[ch];
      } else if (ch < 27) {
        float s = 2.f / (1.f + __expf(-(v + mod_b[ch - 18])));
        mask_out[(((size_t)(b * 9 + (ch - 18))) << 14) + hw] = s;
      }
    }
}

// ---------------- fused deformable-sample + GEMM, double-buffered + TDM ---------
__global__ __launch_bounds__(TPB) void deform_gemm_wmma_kernel(
    const float* __restrict__ src, const float* __restrict__ off,
    const float* __restrict__ msk, const unsigned short* __restrict__ wbf,
    float* __restrict__ out) {
  __shared__ unsigned short As[2][C_CH][40];    // weight tiles, row stride 80B
  __shared__ unsigned short Bs[2][TILE_N][48];  // sampled im2col tiles
  __shared__ int   gIdx[KKONST][TILE_N][4];
  __shared__ float gWt [KKONST][TILE_N][4];

  const int tid   = threadIdx.x;
  const int lane  = tid & 31;
  const int wave  = tid >> 5;
  const int half  = lane >> 4;
  const int lan16 = lane & 15;
  const int j     = tid & 63;           // fixed pixel per thread
  const int klb   = tid >> 6;           // fixed kl residue (0..2), stride 3

  const int b    = blockIdx.x >> 8;
  const int tile = blockIdx.x & 255;
  const int hw0  = tile * TILE_N;
  const int h    = hw0 >> 7;
  const int w0   = hw0 & (WW2 - 1);
  const size_t xbatch = ((size_t)b * C_CH) << 14;
  const int m0 = wave * 16;

  // per-(tap,pixel) bilinear tables, shared by all 96 channels
  for (int e = tid; e < KKONST * TILE_N; e += TPB) {
    int kk = e >> 6;
    int jj = e & 63;
    int w  = w0 + jj;
    int ky = kk / 3, kx = kk - 3 * ky;
    size_t ob = (((size_t)(b * 2 * KKONST + kk * 2)) << 14) + (h << 7) + w;
    float oy = off[ob];
    float ox = off[ob + HW];
    float m  = msk[(((size_t)(b * KKONST + kk)) << 14) + (h << 7) + w];
    float py = (float)(h - 1 + ky) + oy;
    float px = (float)(w - 1 + kx) + ox;
    float y0f = floorf(py), x0f = floorf(px);
    float dy = py - y0f, dx = px - x0f;
    int y0 = (int)y0f, x0 = (int)x0f;
#pragma unroll
    for (int q = 0; q < 4; ++q) {
      int yy = y0 + (q >> 1);
      int xx = x0 + (q & 1);
      bool ok = (yy >= 0) && (yy < HH) && (xx >= 0) && (xx < WW2);
      int yc = min(max(yy, 0), HH - 1);
      int xc = min(max(xx, 0), WW2 - 1);
      float wy = (q >> 1) ? dy : (1.f - dy);
      float wx = (q & 1) ? dx : (1.f - dx);
      gIdx[kk][jj][q] = yc * WW2 + xc;
      gWt [kk][jj][q] = ok ? wy * wx * m : 0.f;
    }
  }

  // ---- tile helpers ----
  auto gatherB = [&](int kb, float* vbuf) {
    int k  = kb + klb;
    int c  = k / 9;                     // one division per tile
    int kk = k - 9 * c;
#pragma unroll
    for (int i = 0; i < NELEM; ++i) {   // kl = 3*i + klb
      int kl = 3 * i + klb;
      if (kl >= KSTEP) break;
      const float* xb = src + xbatch + ((size_t)c << 14);
      const int*   gi = gIdx[kk][j];
      const float* gw = gWt[kk][j];
      vbuf[i] = gw[0] * xb[gi[0]] + gw[1] * xb[gi[1]] +
                gw[2] * xb[gi[2]] + gw[3] * xb[gi[3]];
      kk += 3; if (kk >= 9) { kk -= 9; ++c; }
    }
  };
  auto storeB = [&](int bufi, const float* vbuf) {
#pragma unroll
    for (int i = 0; i < NELEM; ++i) {
      int kl = 3 * i + klb;
      if (kl >= KSTEP) break;
      Bs[bufi][j][kl] = f32_to_bf16(vbuf[i]);
    }
  };

#if USE_TDM
  // 2-D TDM descriptor: 96x32 bf16 tile out of the 96x864 weight tensor,
  // LDS rows padded 64B->80B via pad_interval=16 DW, pad_amount=4 DW.
  auto tdm_issueA = [&](int bufi, int kb) {
    if (wave == 0) {
      unsigned lds_off = (unsigned)(unsigned long long)(&As[bufi][0][0]);
      unsigned long long ga = (unsigned long long)(const unsigned short*)(wbf + kb);
      v4u g0;
      g0[0] = 1u;                                        // count=1
      g0[1] = lds_off;                                   // lds_addr
      g0[2] = (unsigned)(ga & 0xFFFFFFFFu);              // global_addr[31:0]
      g0[3] = (unsigned)((ga >> 32) & 0x1FFFFFFu) | (2u << 30);  // [56:32] | type=2
      v8i g1;
      g1[0] = (int)((1u << 16)          // data_size = 2B
                  | (1u << 20)          // pad_enable
                  | (3u << 22)          // pad_interval: 16 DWORDs
                  | (3u << 25));        // pad_amount: 4 DWORDs
      g1[1] = (int)((864u & 0xFFFFu) << 16);             // tensor_dim0[15:0]
      g1[2] = (int)((864u >> 16) | (96u << 16));         // td0[31:16] | td1[15:0]
      g1[3] = (int)(32u << 16);                          // tile_dim0 = 32
      g1[4] = (int)96;                                   // tile_dim1 = 96
      g1[5] = (int)864;                                  // tensor_dim0_stride lo
      g1[6] = 0;
      g1[7] = 0;
      v4i gz = {};
#if TDM_6ARG
      v8i gz8 = {};
      __builtin_amdgcn_tensor_load_to_lds(g0, g1, gz, gz, gz8, 0);
#else
      __builtin_amdgcn_tensor_load_to_lds(g0, g1, gz, gz, 0);
#endif
    }
  };
#else
  auto stageA = [&](int bufi, int kb) {
    // 96x32 bf16 = 768 8-byte chunks, 4 per thread
    for (int i = tid; i < 768; i += TPB) {
      int m  = i >> 3;
      int kq = (i & 7) * 4;
      *(unsigned long long*)&As[bufi][m][kq] =
          *(const unsigned long long*)&wbf[(size_t)m * KTOT + kb + kq];
      if (kq == 0 && kb + KSTEP < KTOT)
        __builtin_prefetch(&wbf[(size_t)m * KTOT + kb + KSTEP], 0, 1);
    }
  };
#endif

  // ---- prologue: stage tile 0 ----
  float vbuf[NELEM];
#if USE_TDM
  tdm_issueA(0, 0);
#else
  stageA(0, 0);
#endif
  __syncthreads();            // gIdx/gWt tables ready before gather
  gatherB(0, vbuf);
  storeB(0, vbuf);
#if USE_TDM
  if (wave == 0) __builtin_amdgcn_s_wait_tensorcnt(0);
#endif
  __syncthreads();

  v8f acc[4] = {};
  int cur = 0;

  for (int k0 = 0; k0 < KTOT; k0 += KSTEP) {
    const int nxt = cur ^ 1;
    const int knext = k0 + KSTEP;
    const bool hn = knext < KTOT;

#if USE_TDM
    if (hn) tdm_issueA(nxt, knext);   // DMA overlaps WMMA below
#endif
    if (hn) gatherB(knext, vbuf);     // vmem gathers overlap WMMA below

    // ---- WMMA on current tile ----
    v16us araw;
    {
      int m = m0 + lan16;
#pragma unroll
      for (int i = 0; i < 8; ++i) {
        araw[i]     = As[cur][m][half * 8 + i];
        araw[8 + i] = As[cur][m][16 + half * 8 + i];
      }
    }
    v16bf afrag = __builtin_bit_cast(v16bf, araw);
#pragma unroll
    for (int nb = 0; nb < 4; ++nb) {
      int n = nb * 16 + lan16;
      v16us braw;
#pragma unroll
      for (int i = 0; i < 16; ++i) braw[i] = Bs[cur][n][((i >> 3) << 4) + half * 8 + (i & 7)];
      v16bf bfrag = __builtin_bit_cast(v16bf, braw);
      acc[nb] = __builtin_amdgcn_wmma_f32_16x16x32_bf16(
          false, afrag, false, bfrag, (short)0, acc[nb], false, false);
    }

#if !USE_TDM
    if (hn) stageA(nxt, knext);
#endif
    if (hn) storeB(nxt, vbuf);
#if USE_TDM
    if (hn && wave == 0) __builtin_amdgcn_s_wait_tensorcnt(0);
#endif
    __syncthreads();
    cur = nxt;
  }

  // Epilogue: D layout — VGPR r: lanes0-15 M=r, lanes16-31 M=r+8; N = lane%16.
#pragma unroll
  for (int nb = 0; nb < 4; ++nb) {
    int hw = hw0 + nb * 16 + lan16;
#pragma unroll
    for (int r = 0; r < 8; ++r) {
      int ch = m0 + r + (half << 3);
      out[(((size_t)(b * C_CH + ch)) << 14) + hw] = acc[nb][r];
    }
  }
}

// ------------------------------------------------ per-channel batchnorm stats
__global__ void bn_stats_kernel(const float* __restrict__ x,
                                float* __restrict__ mean,
                                float* __restrict__ rstd) {
  __shared__ float s1[256], s2[256];
  int ch  = blockIdx.x;
  int tid = threadIdx.x;
  float a = 0.f, a2 = 0.f;
  for (int i = tid; i < BB * HW; i += 256) {
    int bb = i >> 14;
    int hw = i & (HW - 1);
    float v = x[(((size_t)(bb * C_CH + ch)) << 14) + hw];
    a += v;
    a2 += v * v;
  }
  s1[tid] = a; s2[tid] = a2;
  __syncthreads();
  for (int s = 128; s > 0; s >>= 1) {
    if (tid < s) { s1[tid] += s1[tid + s]; s2[tid] += s2[tid + s]; }
    __syncthreads();
  }
  if (tid == 0) {
    float inv = 1.f / (float)(BB * HW);
    float mu  = s1[0] * inv;
    float var = s2[0] * inv - mu * mu;
    mean[ch] = mu;
    rstd[ch] = rsqrtf(var + EPS_BN);
  }
}

// ------------------------------- BN scale/shift (+ optional leaky / residual)
__global__ void bn_apply_kernel(const float* __restrict__ x,
                                const float* __restrict__ mean,
                                const float* __restrict__ rstd,
                                const float* __restrict__ g,
                                const float* __restrict__ bta,
                                const float* __restrict__ resid,
                                float* __restrict__ out, int leaky) {
  int i = blockIdx.x * blockDim.x + threadIdx.x;
  if (i >= BB * C_CH * HW) return;
  int ch = (i >> 14) % C_CH;
  float v = (x[i] - mean[ch]) * rstd[ch] * g[ch] + bta[ch];
  if (leaky) v = (v >= 0.f) ? v : NEG_SLOPE * v;
  if (resid) v += resid[i];
  out[i] = v;
}

extern "C" void kernel_launch(void* const* d_in, const int* in_sizes, int n_in,
                              void* d_out, int out_size, void* d_ws, size_t ws_size,
                              hipStream_t stream) {
  const float* x       = (const float*)d_in[0];
  const float* d1_offw = (const float*)d_in[1];
  const float* d1_offb = (const float*)d_in[2];
  const float* d1_modw = (const float*)d_in[3];
  const float* d1_modb = (const float*)d_in[4];
  const float* d1_w    = (const float*)d_in[5];
  const float* d2_offw = (const float*)d_in[6];
  const float* d2_offb = (const float*)d_in[7];
  const float* d2_modw = (const float*)d_in[8];
  const float* d2_modb = (const float*)d_in[9];
  const float* d2_w    = (const float*)d_in[10];
  const float* bn_g    = (const float*)d_in[11];
  const float* bn_b    = (const float*)d_in[12];

  char* ws = (char*)d_ws;
  size_t cur = 0;
  auto wsalloc = [&](size_t bytes) -> void* {
    void* p = ws + cur;
    cur = (cur + bytes + 255) & ~(size_t)255;
    return p;
  };
  unsigned short* Abf1 = (unsigned short*)wsalloc((size_t)C_CH * KTOT * 2);
  unsigned short* Abf2 = (unsigned short*)wsalloc((size_t)C_CH * KTOT * 2);
  unsigned short* Aom1 = (unsigned short*)wsalloc((size_t)32 * KTOT * 2);
  unsigned short* Aom2 = (unsigned short*)wsalloc((size_t)32 * KTOT * 2);
  float* offb  = (float*)wsalloc((size_t)BB * 18 * HW * 4);
  float* mskb  = (float*)wsalloc((size_t)BB * 9 * HW * 4);
  float* h1    = (float*)wsalloc((size_t)BB * C_CH * HW * 4);
  float* h1n   = (float*)wsalloc((size_t)BB * C_CH * HW * 4);
  float* mean1 = (float*)wsalloc(C_CH * 4);
  float* rstd1 = (float*)wsalloc(C_CH * 4);
  float* mean2 = (float*)wsalloc(C_CH * 4);
  float* rstd2 = (float*)wsalloc(C_CH * 4);

  const int nw = C_CH * KTOT;           // 82944
  pack_bf16_kernel<<<(nw + 255) / 256, 256, 0, stream>>>(d1_w, Abf1, nw);
  pack_bf16_kernel<<<(nw + 255) / 256, 256, 0, stream>>>(d2_w, Abf2, nw);
  const int nom = 32 * KTOT;            // 27648
  pack_offmask_w_kernel<<<(nom + 255) / 256, 256, 0, stream>>>(d1_offw, d1_modw, Aom1);
  pack_offmask_w_kernel<<<(nom + 255) / 256, 256, 0, stream>>>(d2_offw, d2_modw, Aom2);

  // conv 1
  offmask_gemm_wmma_kernel<<<BB * 256, 128, 0, stream>>>(x, Aom1, d1_offb, d1_modb,
                                                         offb, mskb);
  deform_gemm_wmma_kernel<<<BB * 256, TPB, 0, stream>>>(x, offb, mskb, Abf1, h1);
  bn_stats_kernel<<<C_CH, 256, 0, stream>>>(h1, mean1, rstd1);
  const int ntot = BB * C_CH * HW;      // 6291456
  bn_apply_kernel<<<(ntot + 255) / 256, 256, 0, stream>>>(
      h1, mean1, rstd1, bn_g, bn_b, nullptr, h1n, 1);

  // conv 2 (reuse off/mask buffers and h1 as h2)
  offmask_gemm_wmma_kernel<<<BB * 256, 128, 0, stream>>>(h1n, Aom2, d2_offb, d2_modb,
                                                         offb, mskb);
  deform_gemm_wmma_kernel<<<BB * 256, TPB, 0, stream>>>(h1n, offb, mskb, Abf2, h1);
  bn_stats_kernel<<<C_CH, 256, 0, stream>>>(h1, mean2, rstd2);
  bn_apply_kernel<<<(ntot + 255) / 256, 256, 0, stream>>>(
      h1, mean2, rstd2, bn_g, bn_b, x, (float*)d_out, 0);
}
